// MaskedMHA_44590350467481
// MI455X (gfx1250) — matrix operand order repaired and checked by reading, not compiled, not run on hardware
//
#include <hip/hip_runtime.h>

#define S_LEN 2048
#define BATCH 4
#define DMODEL 512
#define NHEAD 8
#define DK 64
#define NTOK (S_LEN * BATCH)
#define NEGINF (-1.0e10f)

typedef __bf16 bf16_t;
typedef __attribute__((ext_vector_type(16))) __bf16 v16bf;
typedef __attribute__((ext_vector_type(8)))  __bf16 bf16x8;
typedef __attribute__((ext_vector_type(8)))  float  v8f;

// D = A(16x32 bf16) x B(32x16 bf16) + C(16x16 f32)
#define WMMA_BF16(A, B, C) \
  __builtin_amdgcn_wmma_f32_16x16x32_bf16(false, (A), false, (B), (short)0, (C), false, false)

// ---------------------------------------------------------------------------
// Prologue A: elementwise f32 -> bf16 for the activations (S,B,D).
// ---------------------------------------------------------------------------
__global__ __launch_bounds__(256) void cvt_x_kernel(const float* __restrict__ x,
                                                    bf16_t* __restrict__ o) {
  const size_t i = ((size_t)blockIdx.x * 256 + threadIdx.x) * 4;
  const float4 v = *(const float4*)(x + i);
  o[i + 0] = (bf16_t)v.x;
  o[i + 1] = (bf16_t)v.y;
  o[i + 2] = (bf16_t)v.z;
  o[i + 3] = (bf16_t)v.w;
}

// ---------------------------------------------------------------------------
// Prologue B: transpose + convert DxD weight to bf16 column-major wT[n][k],
// so GEMM B-fragments become contiguous 32B loads. LDS-tiled for coalescing.
// ---------------------------------------------------------------------------
__global__ __launch_bounds__(256) void transpose_cvt_kernel(
    const float* __restrict__ w0, const float* __restrict__ w1,
    const float* __restrict__ w2, const float* __restrict__ w3,
    bf16_t* __restrict__ o0, bf16_t* __restrict__ o1,
    bf16_t* __restrict__ o2, bf16_t* __restrict__ o3)
{
  __shared__ float tile[32][33];
  const float* w; bf16_t* o;
  switch (blockIdx.z) {
    case 0:  w = w0; o = o0; break;
    case 1:  w = w1; o = o1; break;
    case 2:  w = w2; o = o2; break;
    default: w = w3; o = o3; break;
  }
  const int tx = threadIdx.x, ty = threadIdx.y;       // block (32,8)
  const int n = blockIdx.x * 32 + tx;                 // column of w
  const int k0 = blockIdx.y * 32;                     // row base of w
#pragma unroll
  for (int j = ty; j < 32; j += 8)
    tile[j][tx] = w[(size_t)(k0 + j) * DMODEL + n];
  __syncthreads();
  const int k = k0 + tx;                              // contiguous in output
  const int n0 = blockIdx.x * 32;
#pragma unroll
  for (int j = ty; j < 32; j += 8)
    o[(size_t)(n0 + j) * DMODEL + k] = (bf16_t)tile[tx][j];
}

// A-fragment (16x32 bf16) from a contiguous bf16 row (contiguous k = 16h + s layout).
__device__ __forceinline__ v16bf load_a_bf16(const bf16_t* __restrict__ p) {
  v16bf a;
#pragma unroll
  for (int i = 0; i < 8; ++i) { a[i] = p[i]; a[i + 8] = p[8 + i]; }
  return a;
}

// ---------------------------------------------------------------------------
// Kernel 1: fused QKV projection, software-pipelined (next-iter fragments
// issued before current-iter WMMAs so waits retire at loadcnt > 0).
// Q,K stored [b,h,s,dk]; V stored transposed [b,h,dk,s].
// ---------------------------------------------------------------------------
__global__ __launch_bounds__(128) void qkv_proj_kernel(
    const bf16_t* __restrict__ xbf,
    const bf16_t* __restrict__ wqT, const float* __restrict__ bq,
    const bf16_t* __restrict__ wkT, const float* __restrict__ bk,
    const bf16_t* __restrict__ wvT, const float* __restrict__ bv,
    bf16_t* __restrict__ qws, bf16_t* __restrict__ kws, bf16_t* __restrict__ vtws)
{
  const int lane = threadIdx.x & 31;
  const int wave = threadIdx.x >> 5;
  const int rowt = blockIdx.x;                 // 0..511 (token tiles)
  const int colt = blockIdx.y * 4 + wave;      // 0..31  (column tiles)
  const int nloc = lane & 15;
  const int hi   = lane >> 4;
  const int abase = hi ? 16 : 0;
  const int kb2   = hi ? 16 : 0;
  const int n     = colt * 16 + nloc;          // output column (0..511)
  const bf16_t* arow = xbf + (size_t)(rowt * 16 + nloc) * DMODEL + abase;
  const bf16_t* bq_p = wqT + (size_t)n * DMODEL + kb2;
  const bf16_t* bk_p = wkT + (size_t)n * DMODEL + kb2;
  const bf16_t* bv_p = wvT + (size_t)n * DMODEL + kb2;

  v8f cq = {}; v8f ck = {}; v8f cv = {};
  // pipeline prologue
  v16bf a  = load_a_bf16(arow);
  v16bf b0 = *(const v16bf*)(bq_p);
  v16bf b1 = *(const v16bf*)(bk_p);
  v16bf b2 = *(const v16bf*)(bv_p);
#pragma unroll 5
  for (int kk = 32; kk < DMODEL; kk += 32) {
    v16bf an  = load_a_bf16(arow + kk);
    v16bf b0n = *(const v16bf*)(bq_p + kk);
    v16bf b1n = *(const v16bf*)(bk_p + kk);
    v16bf b2n = *(const v16bf*)(bv_p + kk);
    cq = WMMA_BF16(a, b0, cq);
    ck = WMMA_BF16(a, b1, ck);
    cv = WMMA_BF16(a, b2, cv);
    a = an; b0 = b0n; b1 = b1n; b2 = b2n;
  }
  cq = WMMA_BF16(a, b0, cq);
  ck = WMMA_BF16(a, b1, ck);
  cv = WMMA_BF16(a, b2, cv);

  const int h = n >> 6, dk = n & 63;
  const float biasq = bq[n], biask = bk[n], biasv = bv[n];
#pragma unroll
  for (int r = 0; r < 8; ++r) {
    const int t = rowt * 16 + r + hi * 8;      // token index = s*B + b
    const int s = t >> 2, b = t & 3;
    const size_t bh = (size_t)(b * NHEAD + h);
    qws[(bh * S_LEN + s) * DK + dk]  = (bf16_t)(cq[r] + biasq);
    kws[(bh * S_LEN + s) * DK + dk]  = (bf16_t)(ck[r] + biask);
    vtws[(bh * DK + dk) * S_LEN + s] = (bf16_t)(cv[r] + biasv);
  }
}

// ---------------------------------------------------------------------------
// Kernel 2: flash attention. One 16-query tile per wave, 32-key blocks,
// online softmax with causal + key-pad masking. P transposed via LDS.
// Pipelined: V fragments issued before softmax; next K block preloaded.
// ---------------------------------------------------------------------------
__global__ __launch_bounds__(128) void flash_attn_kernel(
    const bf16_t* __restrict__ qws, const bf16_t* __restrict__ kws,
    const bf16_t* __restrict__ vtws, const unsigned char* __restrict__ padmask,
    bf16_t* __restrict__ aws)
{
  __shared__ __bf16 p_lds[4][16 * 32];         // one 16x32 P tile per wave

  const int lane = threadIdx.x & 31;
  const int wave = threadIdx.x >> 5;
  const int nloc = lane & 15;
  const int hi   = lane >> 4;
  const int qt = blockIdx.x * 4 + wave;        // 0..127 query tiles
  const int bh = blockIdx.y;                   // 0..31
  const int b  = bh >> 3;
  const int s_base = qt * 16;

  const bf16_t* qbase = qws  + (size_t)bh * S_LEN * DK;
  const bf16_t* kbase = kws  + (size_t)bh * S_LEN * DK;
  const bf16_t* vbase = vtws + (size_t)bh * DK * S_LEN;

  const int abase = hi ? 16 : 0;
  const bf16_t* qrow = qbase + (size_t)(s_base + nloc) * DK + abase;
  v16bf aq0 = load_a_bf16(qrow);
  v16bf aq1 = load_a_bf16(qrow + 32);

  float mrow[8], lrow[8];
#pragma unroll
  for (int r = 0; r < 8; ++r) { mrow[r] = -3.0e38f; lrow[r] = 0.0f; }
  v8f acc0 = {}; v8f acc1 = {}; v8f acc2 = {}; v8f acc3 = {};

  // K fragment base for this lane: sub-tile t (0/1) at +t*16*DK, dk half at +32
  const bf16_t* krow = kbase + (size_t)nloc * DK + (hi ? 16 : 0);
  v16bf kf00 = *(const v16bf*)(krow);
  v16bf kf01 = *(const v16bf*)(krow + 32);
  v16bf kf10 = *(const v16bf*)(krow + 16 * DK);
  v16bf kf11 = *(const v16bf*)(krow + 16 * DK + 32);

  const int nblk = (s_base + 15) / 32 + 1;     // causal bound
  for (int blk = 0; blk < nblk; ++blk) {
    const int kb = blk * 32;
    // prefetch two blocks ahead (gfx1250 global_prefetch_b8)
    __builtin_prefetch(kbase + (size_t)(kb + 64 + nloc) * DK, 0, 1);

    // ---- V fragments for this block: issue early, consume after softmax ----
    const bf16_t* vcol = vbase + (size_t)nloc * S_LEN + kb + (hi ? 16 : 0);
    v16bf vf0 = *(const v16bf*)(vcol + (size_t)0 * 16 * S_LEN);
    v16bf vf1 = *(const v16bf*)(vcol + (size_t)1 * 16 * S_LEN);
    v16bf vf2 = *(const v16bf*)(vcol + (size_t)2 * 16 * S_LEN);
    v16bf vf3 = *(const v16bf*)(vcol + (size_t)3 * 16 * S_LEN);

    // ---- S = Q x K^T for two 16-key sub-tiles ----
    v8f sc0 = {};
    sc0 = WMMA_BF16(aq0, kf00, sc0);
    sc0 = WMMA_BF16(aq1, kf01, sc0);
    v8f sc1 = {};
    sc1 = WMMA_BF16(aq0, kf10, sc1);
    sc1 = WMMA_BF16(aq1, kf11, sc1);

    // ---- preload next K block (clamped on last iter) ----
    const int kbn = (blk + 1 < nblk) ? kb + 32 : kb;
    const bf16_t* krn = krow + (size_t)kbn * DK;
    kf00 = *(const v16bf*)(krn);
    kf01 = *(const v16bf*)(krn + 32);
    kf10 = *(const v16bf*)(krn + 16 * DK);
    kf11 = *(const v16bf*)(krn + 16 * DK + 32);

    // ---- scale + causal/pad mask ----
    const int key0 = kb + nloc;
    const int key1 = key0 + 16;
    const bool pad0 = padmask[(size_t)key0 * BATCH + b] != 0;
    const bool pad1 = padmask[(size_t)key1 * BATCH + b] != 0;
#pragma unroll
    for (int r = 0; r < 8; ++r) {
      const int row = s_base + r + hi * 8;
      float v0 = sc0[r] * 0.125f;              // 1/sqrt(64)
      float v1 = sc1[r] * 0.125f;
      if (key0 > row || pad0) v0 = NEGINF;
      if (key1 > row || pad1) v1 = NEGINF;
      sc0[r] = v0; sc1[r] = v1;
    }

    // ---- online softmax (row = per-VGPR, reduce across 16-lane half) ----
    float scale_r[8];
#pragma unroll
    for (int r = 0; r < 8; ++r) {
      float cand = fmaxf(sc0[r], sc1[r]);
#pragma unroll
      for (int off = 1; off < 16; off <<= 1)
        cand = fmaxf(cand, __shfl_xor(cand, off, 32));
      const float mnew = fmaxf(mrow[r], cand);
      const float sold = __expf(mrow[r] - mnew);
      const float p0 = __expf(sc0[r] - mnew);
      const float p1 = __expf(sc1[r] - mnew);
      float psum = p0 + p1;
#pragma unroll
      for (int off = 1; off < 16; off <<= 1)
        psum += __shfl_xor(psum, off, 32);
      lrow[r] = lrow[r] * sold + psum;
      mrow[r] = mnew;
      scale_r[r] = sold;
      const int m = r + hi * 8;                // C-layout -> row-major P tile
      p_lds[wave][m * 32 + nloc]      = (bf16_t)p0;
      p_lds[wave][m * 32 + 16 + nloc] = (bf16_t)p1;
    }
    asm volatile("" ::: "memory");             // order LDS write -> read (same wave, DS in-order)

    // ---- rescale O, read P as A-fragment, P x V ----
#pragma unroll
    for (int r = 0; r < 8; ++r) {
      acc0[r] *= scale_r[r]; acc1[r] *= scale_r[r];
      acc2[r] *= scale_r[r]; acc3[r] *= scale_r[r];
    }
    const bf16x8* prow = (const bf16x8*)&p_lds[wave][nloc * 32];
    bf16x8 plo = prow[2 * hi];
    bf16x8 phi = prow[2 * hi + 1];
    v16bf pf;
#pragma unroll
    for (int i = 0; i < 8; ++i) { pf[i] = plo[i]; pf[i + 8] = phi[i]; }

    acc0 = WMMA_BF16(pf, vf0, acc0);
    acc1 = WMMA_BF16(pf, vf1, acc1);
    acc2 = WMMA_BF16(pf, vf2, acc2);
    acc3 = WMMA_BF16(pf, vf3, acc3);
    asm volatile("" ::: "memory");
  }

  // ---- normalize and store attn in (S,B,D) bf16 layout ----
  const int h = bh & 7;
#pragma unroll
  for (int r = 0; r < 8; ++r) {
    const float rl = 1.0f / lrow[r];
    const int srow = s_base + r + hi * 8;
    bf16_t* orow = aws + (size_t)(srow * BATCH + b) * DMODEL + h * DK + nloc;
    orow[0]  = (bf16_t)(acc0[r] * rl);
    orow[16] = (bf16_t)(acc1[r] * rl);
    orow[32] = (bf16_t)(acc2[r] * rl);
    orow[48] = (bf16_t)(acc3[r] * rl);
  }
}

// ---------------------------------------------------------------------------
// Kernel 3: output projection. attn(bf16) @ woT(bf16) + bo -> f32 out.
// Software-pipelined like kernel 1.
// ---------------------------------------------------------------------------
__global__ __launch_bounds__(128) void out_proj_kernel(
    const bf16_t* __restrict__ A, const bf16_t* __restrict__ woT,
    const float* __restrict__ bo, float* __restrict__ out)
{
  const int lane = threadIdx.x & 31;
  const int wave = threadIdx.x >> 5;
  const int rowt = blockIdx.x;
  const int colt = blockIdx.y * 4 + wave;
  const int nloc = lane & 15;
  const int hi   = lane >> 4;
  const int abase = hi ? 16 : 0;
  const int kb2   = hi ? 16 : 0;
  const int n = colt * 16 + nloc;
  const bf16_t* arow = A + (size_t)(rowt * 16 + nloc) * DMODEL + abase;
  const bf16_t* bp   = woT + (size_t)n * DMODEL + kb2;

  v8f c = {};
  v16bf a = load_a_bf16(arow);
  v16bf bb = *(const v16bf*)(bp);
#pragma unroll 5
  for (int kk = 32; kk < DMODEL; kk += 32) {
    v16bf an = load_a_bf16(arow + kk);
    v16bf bn = *(const v16bf*)(bp + kk);
    c = WMMA_BF16(a, bb, c);
    a = an; bb = bn;
  }
  c = WMMA_BF16(a, bb, c);

  const float bias = bo[n];
#pragma unroll
  for (int r = 0; r < 8; ++r) {
    const int t = rowt * 16 + r + hi * 8;
    out[(size_t)t * DMODEL + n] = c[r] + bias;
  }
}

extern "C" void kernel_launch(void* const* d_in, const int* in_sizes, int n_in,
                              void* d_out, int out_size, void* d_ws, size_t ws_size,
                              hipStream_t stream) {
  const float* emb = (const float*)d_in[0];
  const unsigned char* mask = (const unsigned char*)d_in[1];  // (S,B) bool
  const float* wq = (const float*)d_in[2]; const float* bq = (const float*)d_in[3];
  const float* wk = (const float*)d_in[4]; const float* bk = (const float*)d_in[5];
  const float* wv = (const float*)d_in[6]; const float* bv = (const float*)d_in[7];
  const float* wo = (const float*)d_in[8]; const float* bo = (const float*)d_in[9];

  const size_t big = (size_t)NTOK * DMODEL;   // 4,194,304 elems
  const size_t wsz = (size_t)DMODEL * DMODEL; //   262,144 elems
  bf16_t* xbf  = (bf16_t*)d_ws;
  bf16_t* wqT  = xbf  + big;
  bf16_t* wkT  = wqT  + wsz;
  bf16_t* wvT  = wkT  + wsz;
  bf16_t* woT  = wvT  + wsz;
  bf16_t* qws  = woT  + wsz;
  bf16_t* kws  = qws  + big;
  bf16_t* vtws = kws  + big;
  bf16_t* aws  = vtws + big;                  // attn, (S*B, D) bf16

  // Prologue: one-time conversions (removes all f32->bf16 from GEMM loops)
  cvt_x_kernel<<<dim3(big / 1024, 1, 1), dim3(256, 1, 1), 0, stream>>>(emb, xbf);
  transpose_cvt_kernel<<<dim3(16, 16, 4), dim3(32, 8, 1), 0, stream>>>(
      wq, wk, wv, wo, wqT, wkT, wvT, woT);

  dim3 blk(128, 1, 1);
  qkv_proj_kernel<<<dim3(512, 8, 1), blk, 0, stream>>>(xbf, wqT, bq, wkT, bk, wvT, bv,
                                                       qws, kws, vtws);
  flash_attn_kernel<<<dim3(32, 32, 1), blk, 0, stream>>>(qws, kws, vtws, mask, aws);
  out_proj_kernel<<<dim3(512, 8, 1), blk, 0, stream>>>(aws, woT, bo, (float*)d_out);
}
